// Lstm_9569187136054
// MI455X (gfx1250) — matrix operand order repaired and checked by reading, not compiled, not run on hardware
//
#include <hip/hip_runtime.h>
#include <hip/hip_bf16.h>

typedef __attribute__((ext_vector_type(16))) _Float16 v16h;
typedef __attribute__((ext_vector_type(8)))  _Float16 v8h;
typedef __attribute__((ext_vector_type(8)))  float    v8f;

#define WMMA_F32_F16(A, B, C) \
    __builtin_amdgcn_wmma_f32_16x16x32_f16(false, (A), false, (B), (short)0, (C), false, false)

static constexpr int BATCH = 256;
static constexpr int T     = 1024;
static constexpr int D     = 128;
static constexpr int H     = 64;
static constexpr int G     = 256;   // 4*H gate width
static constexpr int OUT   = 10;

// ---------------------------------------------------------------------------
// Kernel 0: pre-swizzle W [D][G] f32 -> f16 WMMA B-fragment layout:
//   Wfrag[((nt*4 + kc)*32 + lane)*16 + e] = W[(kc*32 + (lane>>4)*16 + e)*G + nt*16 + (lane&15)]
// 64KB total; lets the GEMM load B-fragments as contiguous b128 pairs.
// ---------------------------------------------------------------------------
__global__ __launch_bounds__(128) void wfrag_kernel(
    const float* __restrict__ W, _Float16* __restrict__ Wfrag)
{
    const int id   = blockIdx.x * 128 + threadIdx.x;   // 32768 elements
    const int e    = id & 15;
    const int lane = (id >> 4) & 31;
    const int kc   = (id >> 9) & 3;
    const int nt   = id >> 11;
    const int k    = kc * 32 + (lane >> 4) * 16 + e;
    const int col  = nt * 16 + (lane & 15);
    Wfrag[id] = (_Float16)W[(size_t)k * G + col];
}

// ---------------------------------------------------------------------------
// Kernel 1: xw = x@W + bias. One wave per (t, 16-batch-row) m-tile, so the
// output C fragment is EXACTLY the tile the scan consumes; stored in
// fragment layout -> scan reads one v8f per gate. A-fragments (x) loaded
// once per wave as vector loads; B-fragments from pre-swizzled Wfrag.
// ---------------------------------------------------------------------------
__global__ __launch_bounds__(256) void xw_gemm_kernel(
    const float* __restrict__ x, const _Float16* __restrict__ Wfrag,
    const float* __restrict__ bias, float* __restrict__ xwF)
{
    const int wave = blockIdx.x * (blockDim.x >> 5) + (threadIdx.x >> 5);
    const int lane = threadIdx.x & 31;
    const int bt   = wave & 15;   // batch tile (16 rows)
    const int t    = wave >> 4;   // time index
    const int m    = lane & 15;
    const int hl   = lane >> 4;
    const int n    = lane & 15;

    // A fragments: 4 K-chunks of 32 (K = D = 128), f32 -> f16
    const float* xrow = x + ((size_t)(bt * 16 + m) * T + t) * D;
    v16h afrag[4];
    #pragma unroll
    for (int kc = 0; kc < 4; ++kc) {
        const int kb = kc * 32 + hl * 16;
        const v8f lo = *(const v8f*)(xrow + kb);        // 2x global_load_b128
        const v8f hi = *(const v8f*)(xrow + kb + 8);    // 2x global_load_b128
        #pragma unroll
        for (int e = 0; e < 8; ++e) {
            afrag[kc][e]     = (_Float16)lo[e];
            afrag[kc][8 + e] = (_Float16)hi[e];
        }
    }

    #pragma unroll 1
    for (int nt = 0; nt < 16; ++nt) {
        v16h bfrag[4];
        #pragma unroll
        for (int kc = 0; kc < 4; ++kc)
            bfrag[kc] = *(const v16h*)&Wfrag[(size_t)((nt * 4 + kc) * 32 + lane) * 16];
        v8f acc;
        const float bv = bias[nt * 16 + n];
        #pragma unroll
        for (int r = 0; r < 8; ++r) acc[r] = bv;
        #pragma unroll
        for (int kc = 0; kc < 4; ++kc) acc = WMMA_F32_F16(afrag[kc], bfrag[kc], acc);
        // store in scan C-fragment layout
        const size_t obase = (((size_t)t * 16 + bt) * 16 + nt) * 256 + lane * 8;
        *(v8f*)&xwF[obase] = acc;                       // 2x global_store_b128
    }
}

// ---------------------------------------------------------------------------
// Kernel 2: the sequential scan + fused dense/softmax. 16 blocks x 4 waves.
// Wave w owns gates {i,f,g,o} x hcols [16w,16w+16) -> c/h update wave-local.
// U lives in registers as B-fragments; h round-trips through double-buffered
// LDS (one barrier/step); xw fragment loads for t+1 are issued at the top of
// the step to overlap the WMMA chain, elementwise update, and barrier.
// ---------------------------------------------------------------------------
__global__ __launch_bounds__(128) void lstm_scan_fused_kernel(
    const float* __restrict__ xwF, const float* __restrict__ U,
    const float* __restrict__ Wd, const float* __restrict__ bd,
    float* __restrict__ out)
{
    __shared__ _Float16 hbuf[2][16 * H];   // double-buffered h tile (f16)
    __shared__ float    hf[16 * H];        // final h (f32) for fused dense

    const int wv   = threadIdx.x >> 5;     // 0..3 -> hcol block
    const int lane = threadIdx.x & 31;
    const int m    = lane & 15;
    const int hl   = lane >> 4;
    const int n    = lane & 15;

    // U B-fragments: [gate][K-chunk], resident all 1024 steps
    v16h ufrag[4][2];
    #pragma unroll
    for (int gi = 0; gi < 4; ++gi) {
        const int col0 = gi * 64 + wv * 16;
        #pragma unroll
        for (int kc = 0; kc < 2; ++kc) {
            const int kb = kc * 32 + hl * 16;
            #pragma unroll
            for (int e = 0; e < 16; ++e)
                ufrag[gi][kc][e] = (_Float16)U[(size_t)(kb + e) * G + col0 + n];
        }
    }

    for (int i = threadIdx.x; i < 16 * H; i += blockDim.x)
        hbuf[0][i] = (_Float16)0.0f;
    __syncthreads();

    v8f cst;
    #pragma unroll
    for (int r = 0; r < 8; ++r) cst[r] = 0.0f;
    float hreg[8];

    // xw fragment base for this (block, wave, lane)
    const size_t zstep = (size_t)16 * 16 * 256;   // per-t stride in xwF
    const float* zbase = xwF + ((size_t)blockIdx.x * 16) * 256 + lane * 8;

    // preload t = 0
    v8f zc[4];
    #pragma unroll
    for (int gi = 0; gi < 4; ++gi)
        zc[gi] = *(const v8f*)(zbase + (size_t)(gi * 4 + wv) * 256);

    #pragma unroll 1
    for (int t = 0; t < T; ++t) {
        const int cur = t & 1, nxt = cur ^ 1;

        // issue next step's fragment loads early (overlap WMMA + barrier)
        const int tn = (t + 1 < T) ? (t + 1) : t;
        v8f zn[4];
        #pragma unroll
        for (int gi = 0; gi < 4; ++gi)
            zn[gi] = *(const v8f*)(zbase + (size_t)tn * zstep +
                                   (size_t)(gi * 4 + wv) * 256);

        // A fragments of h from LDS: two 16B contiguous runs per fragment
        v16h af[2];
        #pragma unroll
        for (int kc = 0; kc < 2; ++kc) {
            const _Float16* hb = &hbuf[cur][m * H + kc * 32 + hl * 16];
            const v8h lo = *(const v8h*)hb;
            const v8h hi = *(const v8h*)(hb + 8);
            #pragma unroll
            for (int e = 0; e < 8; ++e) { af[kc][e] = lo[e]; af[kc][8 + e] = hi[e]; }
        }

        // z += h @ U  (K = 64 -> 2 chained WMMAs per gate)
        #pragma unroll
        for (int gi = 0; gi < 4; ++gi) {
            zc[gi] = WMMA_F32_F16(af[0], ufrag[gi][0], zc[gi]);
            zc[gi] = WMMA_F32_F16(af[1], ufrag[gi][1], zc[gi]);
        }

        // gates + state update (wave-local: all 4 gates for these hcols)
        #pragma unroll
        for (int r = 0; r < 8; ++r) {
            const float iv = fminf(fmaxf(0.2f * zc[0][r] + 0.5f, 0.0f), 1.0f);
            const float fv = fminf(fmaxf(0.2f * zc[1][r] + 0.5f, 0.0f), 1.0f);
            const float gv = fmaxf(zc[2][r], 0.0f);
            const float ov = fminf(fmaxf(0.2f * zc[3][r] + 0.5f, 0.0f), 1.0f);
            const float cv = fv * cst[r] + iv * gv;
            cst[r] = cv;
            const float hv = ov * fmaxf(cv, 0.0f);
            hreg[r] = hv;
            hbuf[nxt][(r + 8 * hl) * H + wv * 16 + n] = (_Float16)hv;
        }
        __syncthreads();

        #pragma unroll
        for (int gi = 0; gi < 4; ++gi) zc[gi] = zn[gi];
    }

    // fused dense + softmax on this block's 16 batch rows (f32 h from regs)
    #pragma unroll
    for (int r = 0; r < 8; ++r)
        hf[(r + 8 * hl) * H + wv * 16 + n] = hreg[r];
    __syncthreads();

    if (threadIdx.x < 16) {
        const int row = threadIdx.x;
        float logit[OUT];
        float mx = -1e30f;
        #pragma unroll
        for (int j = 0; j < OUT; ++j) {
            float s = bd[j];
            #pragma unroll
            for (int k = 0; k < H; ++k) s += hf[row * H + k] * Wd[k * OUT + j];
            logit[j] = s;
            mx = fmaxf(mx, s);
        }
        float denom = 0.0f;
        #pragma unroll
        for (int j = 0; j < OUT; ++j) { logit[j] = expf(logit[j] - mx); denom += logit[j]; }
        const float inv = 1.0f / denom;
        #pragma unroll
        for (int j = 0; j < OUT; ++j)
            out[(blockIdx.x * 16 + row) * OUT + j] = logit[j] * inv;
    }
}

extern "C" void kernel_launch(void* const* d_in, const int* in_sizes, int n_in,
                              void* d_out, int out_size, void* d_ws, size_t ws_size,
                              hipStream_t stream) {
    const float* x    = (const float*)d_in[0];
    const float* W    = (const float*)d_in[1];
    const float* U    = (const float*)d_in[2];
    const float* bias = (const float*)d_in[3];
    const float* Wd   = (const float*)d_in[4];
    const float* bd   = (const float*)d_in[5];
    float* out = (float*)d_out;

    float*    xwF   = (float*)d_ws;                          // 256 MB fragments
    _Float16* Wfrag = (_Float16*)((char*)d_ws +
                        (size_t)T * BATCH * G * sizeof(float)); // 64 KB

    wfrag_kernel<<<256, 128, 0, stream>>>(W, Wfrag);
    // 1024 t x 16 batch-tiles = 16384 waves, 8 per 256-thread block
    xw_gemm_kernel<<<2048, 256, 0, stream>>>(x, Wfrag, bias, xwF);
    lstm_scan_fused_kernel<<<16, 128, 0, stream>>>(xwF, U, Wd, bd, out);
}
